// InEKFSO2_23622320128494
// MI455X (gfx1250) — matrix operand order repaired and checked by reading, not compiled, not run on hardware
//
#include <hip/hip_runtime.h>
#include <hip/hip_bf16.h>
#include <stdint.h>

// ---------------------------------------------------------------------------
// Problem constants (from reference): B=4, T=4096, H=16, NB=32, D=1024, HID=128
// ---------------------------------------------------------------------------
#define BT      16384      // B*T rows
#define DIM     1024       // D
#define HIDW    128        // HID
#define HNB     512        // H*NB output cols
#define REGION  8388608u   // B*T*H*NB  (size of one output tensor region)

// packed-weight dword offsets inside d_ws
#define W1M_OFF 0u          // 32 ktiles * 8 ntiles * 256 dwords = 65536
#define W1G_OFF 65536u
#define W2M_OFF 131072u     // 4 ktiles * 32 ntiles * 256 = 32768
#define W2G_OFF 163840u
#define WPK_TOT 196608u     // total dwords (786 KB)

typedef __attribute__((ext_vector_type(16))) _Float16 v16h;
typedef __attribute__((ext_vector_type(8)))  float    v8f;

union Frag {
    v16h h;
    uint4 q[2];
};

__device__ __forceinline__ uint32_t pack2h(float a, float b) {
    union { _Float16 h[2]; uint32_t u; } p;
    p.h[0] = (_Float16)a;
    p.h[1] = (_Float16)b;
    return p.u;
}

__device__ __forceinline__ unsigned short f16bits(float a) {
    union { _Float16 h; unsigned short u; } p;
    p.h = (_Float16)a;
    return p.u;
}

__device__ __forceinline__ float gelu_exact(float x) {
    return 0.5f * x * (1.0f + erff(x * 0.70710678118654752f));
}

// ---------------------------------------------------------------------------
// Kernel 1: repack f32 weights into CDNA5 WMMA B-fragment order, f16.
// B-matrix layout (16-bit, 32x16 per 16x16x32 WMMA):
//   lane L: column n = L&15, K group base kg = (L>>4)*8
//   VGPR v<4 holds K = kg+2v,+1 ; v>=4 holds K = 16+kg+2(v-4),+1
// Packed dword index: ((kt*NT + nt)*32 + lane)*8 + v
// ---------------------------------------------------------------------------
__global__ __launch_bounds__(256) void prep_weights(
    const float* __restrict__ mw1, const float* __restrict__ mw2,
    const float* __restrict__ gw1, const float* __restrict__ gw2,
    uint32_t* __restrict__ wpk)
{
    uint32_t i = blockIdx.x * 256u + threadIdx.x;
    if (i >= WPK_TOT) return;

    const float* W;
    uint32_t local, NT, Ncols;
    if (i < W1G_OFF)      { W = mw1; local = i;            NT = 8;  Ncols = HIDW; }
    else if (i < W2M_OFF) { W = gw1; local = i - W1G_OFF;  NT = 8;  Ncols = HIDW; }
    else if (i < W2G_OFF) { W = mw2; local = i - W2M_OFF;  NT = 32; Ncols = HNB;  }
    else                  { W = gw2; local = i - W2G_OFF;  NT = 32; Ncols = HNB;  }

    uint32_t v    = local & 7u;
    uint32_t lane = (local >> 3) & 31u;
    uint32_t rest = local >> 8;
    uint32_t nt   = rest % NT;
    uint32_t kt   = rest / NT;

    uint32_t n  = nt * 16u + (lane & 15u);
    uint32_t k0 = kt * 32u + ((v >> 2) * 16u) + ((lane >> 4) * 8u) + 2u * (v & 3u);

    float a = W[(size_t)k0 * Ncols + n];
    float b = W[(size_t)(k0 + 1u) * Ncols + n];
    wpk[i] = pack2h(a, b);
}

// ---------------------------------------------------------------------------
// Kernel 2: fused two-MLP forward for one 16-row tile of content_emb.
//   h1  = gelu(X @ W1 + b1)            16x128   (both MLPs)
//   o2  = h1 @ W2 + b2                 16x512
//   z   = pi*tanh(o2_m)  -> out[theta region]   (consumed in-place by scan)
//   g   = sigmoid(o2_g)  -> out[g region]
// 8 waves / block; wave = one 16-col tile of layer-1, four 16-col tiles of
// layer-2 per MLP.  A-fragment layout (16-bit 16x32):
//   lane L: row m = L&15, K group base kg = (L>>4)*8
//   v<4 : K = kg+2v,+1   ; v>=4 : K = 16+kg+2(v-4),+1
// => per chunk, lane reads LDS dwords [base..base+3] and [base+8..base+11].
// ---------------------------------------------------------------------------
__global__ __launch_bounds__(256) void mlp_kernel(
    const float* __restrict__ X,
    const float* __restrict__ mb1, const float* __restrict__ mb2,
    const float* __restrict__ gb1, const float* __restrict__ gb2,
    const uint32_t* __restrict__ wpk,
    float* __restrict__ out)
{
    __shared__ __align__(16) uint32_t       Xs32[16 * 512];        // 16x1024 f16
    __shared__ __align__(16) unsigned short h1s[2][16][HIDW];      // 2 MLPs x 16x128 f16

    const int tid  = threadIdx.x;
    const int lane = tid & 31;
    const int wid  = tid >> 5;
    const int rowBase = blockIdx.x * 16;

    // ---- stage X tile (16 rows x 1024 f32) into LDS as f16 ----
    const float4* Xg = (const float4*)(X + (size_t)rowBase * DIM);
    #pragma unroll
    for (int it = 0; it < 16; ++it) {
        int f4 = tid + it * 256;             // 0..4095 float4s
        float4 xv = Xg[f4];
        Xs32[f4 * 2 + 0] = pack2h(xv.x, xv.y);
        Xs32[f4 * 2 + 1] = pack2h(xv.z, xv.w);
    }
    __syncthreads();

    const int m  = lane & 15;            // A-fragment row
    const int kg = (lane >> 4) * 8;      // A-fragment K group base

    // ---- layer 1: 16x1024 @ 1024x16 per wave, both MLPs ----
    v8f accm = {}; v8f accg = {};
    for (int kc = 0; kc < 32; ++kc) {
        Frag a;
        const uint4* xa = (const uint4*)&Xs32[m * 512 + kc * 16 + (kg >> 1)];
        a.q[0] = xa[0];
        a.q[1] = xa[2];

        Frag bm, bg;
        const uint4* pm = (const uint4*)&wpk[W1M_OFF + (uint32_t)(kc * 8 + wid) * 256u + (uint32_t)lane * 8u];
        bm.q[0] = pm[0]; bm.q[1] = pm[1];
        const uint4* pg = (const uint4*)&wpk[W1G_OFF + (uint32_t)(kc * 8 + wid) * 256u + (uint32_t)lane * 8u];
        bg.q[0] = pg[0]; bg.q[1] = pg[1];

        accm = __builtin_amdgcn_wmma_f32_16x16x32_f16(false, a.h, false, bm.h,
                                                      (short)0, accm, false, false);
        accg = __builtin_amdgcn_wmma_f32_16x16x32_f16(false, a.h, false, bg.h,
                                                      (short)0, accg, false, false);
    }

    // ---- layer-1 epilogue: bias + exact GELU -> LDS h1 (f16) ----
    {
        const int colL = wid * 16 + (lane & 15);
        const float b1m = mb1[colL];
        const float b1g = gb1[colL];
        #pragma unroll
        for (int v = 0; v < 8; ++v) {
            int mm = v + 8 * (lane >> 4);          // C/D layout: VGPR v -> row
            h1s[0][mm][colL] = f16bits(gelu_exact(accm[v] + b1m));
            h1s[1][mm][colL] = f16bits(gelu_exact(accg[v] + b1g));
        }
    }
    __syncthreads();

    // ---- layer 2: 16x128 @ 128x16 tiles; 4 N-tiles per wave per MLP ----
    for (int mlp = 0; mlp < 2; ++mlp) {
        const uint32_t wbase = (mlp == 0) ? W2M_OFF : W2G_OFF;
        const float*   b2    = (mlp == 0) ? mb2 : gb2;
        const uint32_t* h1d  = (const uint32_t*)&h1s[mlp][0][0];   // 64 dwords/row

        for (int jj = 0; jj < 4; ++jj) {
            const int nt = wid * 4 + jj;
            v8f acc = {};
            #pragma unroll
            for (int kc = 0; kc < 4; ++kc) {
                Frag a;
                const uint4* xa = (const uint4*)&h1d[m * 64 + kc * 16 + (kg >> 1)];
                a.q[0] = xa[0];
                a.q[1] = xa[2];
                Frag bw;
                const uint4* pw = (const uint4*)&wpk[wbase + (uint32_t)(kc * 32 + nt) * 256u + (uint32_t)lane * 8u];
                bw.q[0] = pw[0]; bw.q[1] = pw[1];
                acc = __builtin_amdgcn_wmma_f32_16x16x32_f16(false, a.h, false, bw.h,
                                                             (short)0, acc, false, false);
            }
            const int c = nt * 16 + (lane & 15);
            const float bias = b2[c];
            #pragma unroll
            for (int v = 0; v < 8; ++v) {
                int mm = v + 8 * (lane >> 4);
                size_t r = (size_t)(rowBase + mm);
                float val = acc[v] + bias;
                if (mlp == 0) {
                    // z = pi * tanh(val) -> theta region (scan overwrites in place)
                    out[r * HNB + c] = 3.14159265358979323846f * tanhf(val);
                } else {
                    // g = sigmoid(val) -> g region (final output)
                    out[2u * REGION + r * HNB + c] = 1.0f / (1.0f + expf(-val));
                }
            }
        }
    }
}

// ---------------------------------------------------------------------------
// Kernel 3: sequential Kalman scan. 2048 chains (b,h,nb), T=4096 steps each.
// Reads z from the theta region and overwrites it in place; writes sigma.
// atan2(sin d, cos d) == d - 2*pi*rint(d/(2*pi))  (exact wrap to [-pi,pi)).
// Next-step loads are issued before current compute (software pipeline).
// ---------------------------------------------------------------------------
__global__ __launch_bounds__(256) void scan_kernel(
    const float* __restrict__ theta_path,
    const float* __restrict__ logQ,
    const float* __restrict__ logR,
    const float* __restrict__ logS0,
    float* __restrict__ out)
{
    const int tid = blockIdx.x * 256 + threadIdx.x;   // 0..2047
    const int b = tid >> 9;          // /512
    const int c = tid & 511;         // h*NB + nb

    const float Q = expf(logQ[c]);
    const float R = expf(logR[c]);
    float sigma   = expf(logS0[c]);

    const size_t cb = (size_t)b * 4096u * HNB + c;
    const float* tp = theta_path + cb;
    float*       zo = out + cb;               // theta region (holds z on entry)
    float*       so = out + REGION + cb;      // sigma region
    const float* go = out + 2u * REGION + cb; // g region

    float th = 0.0f, prev = 0.0f;
    float tp_c = tp[0], z_c = zo[0], g_c = go[0];

    for (int t = 0; t < 4096; ++t) {
        const size_t off = (size_t)t * HNB;
        // prefetch next step
        float tp_n = 0.0f, z_n = 0.0f, g_n = 0.0f;
        if (t + 1 < 4096) {
            const size_t o2 = off + HNB;
            tp_n = tp[o2]; z_n = zo[o2]; g_n = go[o2];
        }

        const float inc = tp_c - prev;
        prev = tp_c;
        th    += inc;
        sigma += Q;

        const float diff  = z_c - th;
        const float innov = diff - 6.283185307179586477f *
                            rintf(diff * 0.15915494309189533577f);
        const float Reff  = R / (g_c + 1e-4f);
        const float K     = sigma / (sigma + Reff);
        th    = th + K * innov;
        sigma = (1.0f - K) * sigma;

        zo[off] = th;       // overwrite z with theta_hat
        so[off] = sigma;

        tp_c = tp_n; z_c = z_n; g_c = g_n;
    }
}

// ---------------------------------------------------------------------------
// Launch: prep (weights -> f16 fragments) -> fused MLP (WMMA) -> scan
// ---------------------------------------------------------------------------
extern "C" void kernel_launch(void* const* d_in, const int* in_sizes, int n_in,
                              void* d_out, int out_size, void* d_ws, size_t ws_size,
                              hipStream_t stream)
{
    const float* theta_path = (const float*)d_in[0];
    const float* content    = (const float*)d_in[1];
    const float* mw1 = (const float*)d_in[2];
    const float* mb1 = (const float*)d_in[3];
    const float* mw2 = (const float*)d_in[4];
    const float* mb2 = (const float*)d_in[5];
    const float* gw1 = (const float*)d_in[6];
    const float* gb1 = (const float*)d_in[7];
    const float* gw2 = (const float*)d_in[8];
    const float* gb2 = (const float*)d_in[9];
    const float* logQ  = (const float*)d_in[10];
    const float* logR  = (const float*)d_in[11];
    const float* logS0 = (const float*)d_in[12];

    float*    out = (float*)d_out;
    uint32_t* wpk = (uint32_t*)d_ws;

    prep_weights<<<(WPK_TOT + 255u) / 256u, 256, 0, stream>>>(mw1, mw2, gw1, gw2, wpk);
    mlp_kernel<<<BT / 16, 256, 0, stream>>>(content, mb1, mb2, gb1, gb2, wpk, out);
    scan_kernel<<<2048 / 256, 256, 0, stream>>>(theta_path, logQ, logR, logS0, out);
}
